// FastGRNN_7352984010967
// MI455X (gfx1250) — matrix operand order repaired and checked by reading, not compiled, not run on hardware
//
#include <hip/hip_runtime.h>

// Problem sizes (fixed by reference)
#define BB 64
#define SS 512
#define FF 512
#define HH 1024

typedef __attribute__((ext_vector_type(16))) __bf16 v16bf;
typedef __attribute__((ext_vector_type(8)))  float  v8f;

struct __attribute__((aligned(16))) U128 { unsigned x, y, z, w; };
struct __attribute__((aligned(8)))  U64  { unsigned a, b; };
union Frag { U128 u[2]; v16bf v; };

__device__ __forceinline__ unsigned short f2bf(float f) {
    unsigned u = __builtin_bit_cast(unsigned, f);
    unsigned r = u + 0x7FFFu + ((u >> 16) & 1u);   // round-to-nearest-even
    return (unsigned short)(r >> 16);
}

// A fragment: 16x32 bf16, row-major source, row stride ldk (elems).
// lane l: row = l&15 ; K = (l>>4)*8 + {0..7} in VGPR0-3, +16 in VGPR4-7.
__device__ __forceinline__ v16bf load_frag_a(const unsigned short* p0, int ldk, int lane) {
    const unsigned short* p = p0 + (lane & 15) * ldk + ((lane >> 4) << 3);
    Frag f;
    f.u[0] = *(const U128*)(p);
    f.u[1] = *(const U128*)(p + 16);
    return f.v;
}

// B fragment: 32x16 bf16 (KxN), source stored N-major [N][K] (i.e. rows of W/U),
// row stride ldk. lane l: col = l&15 ; K = (l>>4)*8 + {0..7} in VGPR0-3, +16 in VGPR4-7.
__device__ __forceinline__ v16bf load_frag_b(const unsigned short* p0, int ldk, int lane) {
    const unsigned short* p = p0 + (lane & 15) * ldk + ((lane >> 4) << 3);
    Frag f;
    f.u[0] = *(const U128*)(p);
    f.u[1] = *(const U128*)(p + 16);
    return f.v;
}

__device__ __forceinline__ v8f wmma_bf16(v16bf a, v16bf b, v8f c) {
    return __builtin_amdgcn_wmma_f32_16x16x32_bf16(
        false, a, false, b, (short)0, c, false, false);
}

__device__ __forceinline__ float sigmoidf_(float x) {
    return 1.0f / (1.0f + __expf(-x));
}

// ---------------- fp32 -> bf16 conversion (x4 vectorized) ----------------
__global__ __launch_bounds__(256) void cvt_bf16_kernel(
        const float* __restrict__ src, unsigned short* __restrict__ dst, int n4) {
    int i = blockIdx.x * blockDim.x + threadIdx.x;
    if (i >= n4) return;
    float4 f = ((const float4*)src)[i];
    U64 o;
    o.a = (unsigned)f2bf(f.x) | ((unsigned)f2bf(f.y) << 16);
    o.b = (unsigned)f2bf(f.z) | ((unsigned)f2bf(f.w) << 16);
    ((U64*)dst)[i] = o;
}

// ---------------- init h0 = 0, barrier counter = 0 ----------------
__global__ __launch_bounds__(256) void init_kernel(
        float* __restrict__ h32, unsigned short* __restrict__ h16,
        unsigned* __restrict__ counter, int n) {
    int i = blockIdx.x * blockDim.x + threadIdx.x;
    if (i < n) { h32[i] = 0.0f; h16[i] = 0; }
    if (i == 0) *counter = 0u;
}

// ---------------- Phase 1: Wx GEMM -> d_out ----------------
// C[(b*S+s), h] = sum_f inp[b,s,f] * W[h,f]; M=32768, N=1024, K=512.
// Block = 256 thr (8 waves), tile 64M x 64N; wave = 16M x 32N (2 WMMA tiles).
__global__ __launch_bounds__(256) void wx_gemm_kernel(
        const unsigned short* __restrict__ X,   // [B*S, F] bf16
        const unsigned short* __restrict__ Wb,  // [H, F]   bf16
        float* __restrict__ out) {              // [B*S, H] fp32 (= d_out)
    const int lane = threadIdx.x & 31;
    const int wave = threadIdx.x >> 5;
    const int wm = wave & 3;    // 4 M sub-tiles of 16
    const int wn = wave >> 2;   // 2 N sub-tiles of 32
    const long row0 = (long)blockIdx.x * 64 + wm * 16;
    const int  col0 = blockIdx.y * 64 + wn * 32;

    const unsigned short* arow = X + row0 * FF;
    const unsigned short* bp0  = Wb + (long)col0 * FF;
    const unsigned short* bp1  = Wb + (long)(col0 + 16) * FF;

    v8f acc0 = {}; v8f acc1 = {};
#pragma unroll 4
    for (int k = 0; k < FF; k += 32) {
        v16bf a  = load_frag_a(arow + k, FF, lane);
        v16bf b0 = load_frag_b(bp0 + k, FF, lane);
        v16bf b1 = load_frag_b(bp1 + k, FF, lane);
        acc0 = wmma_bf16(a, b0, acc0);
        acc1 = wmma_bf16(a, b1, acc1);
    }
    const int n  = lane & 15;
    const int mo = (lane >> 4) << 3;
#pragma unroll
    for (int r = 0; r < 8; ++r) {
        long m = row0 + mo + r;
        out[m * HH + col0 + n]      = acc0[r];
        out[m * HH + col0 + 16 + n] = acc1[r];
    }
}

// ---------------- Phase 2: persistent recurrent scan ----------------
// 64 blocks: blockIdx.x&3 -> batch tile of 16, blockIdx.x>>2 -> 64-col H slice.
// U slice (64 rows x 1024 K, bf16) lives in LDS for all 512 steps.
#define NBLK 64
#define LDSK (HH + 8)   // pad 8 bf16 per row -> bank-conflict-free b128 reads

__global__ __launch_bounds__(128) void fastgrnn_scan_kernel(
        float* __restrict__ out,               // [B,S,H]: reads wx_t, writes h_t
        float* __restrict__ hid_out,           // [B,H] final hidden
        const unsigned short* __restrict__ Ub, // [H,H] bf16
        const float* __restrict__ bz,
        const float* __restrict__ bh,
        const float* __restrict__ zeta,
        const float* __restrict__ nu,
        float* __restrict__ h32a, float* __restrict__ h32b,
        unsigned short* __restrict__ h16a, unsigned short* __restrict__ h16b,
        unsigned* __restrict__ counter) {
    extern __shared__ unsigned short Ul[];     // [64][LDSK]

    const int b0 = (blockIdx.x & 3) * 16;
    const int n0 = (blockIdx.x >> 2) * 64;

    // Load U slice rows n0..n0+63 into LDS (b128 copies)
    for (int idx = threadIdx.x; idx < 64 * (HH / 8); idx += blockDim.x) {
        int r = idx >> 7;            // / (HH/8) = /128
        int c = (idx & 127) << 3;    // *8
        *(U128*)&Ul[r * LDSK + c] = *(const U128*)&Ub[(long)(n0 + r) * HH + c];
    }
    __syncthreads();

    const int lane  = threadIdx.x & 31;
    const int wave  = threadIdx.x >> 5;        // 4 waves -> 4 N sub-tiles of 16
    const int nloc0 = wave * 16;
    const int ncol0 = n0 + nloc0;

    const float zs = sigmoidf_(zeta[0]);
    const float ns = sigmoidf_(nu[0]);
    const int nlane = lane & 15;
    const int mo    = (lane >> 4) << 3;
    const int ncol  = ncol0 + nlane;
    const float bzv = bz[ncol];
    const float bhv = bh[ncol];

    for (int s = 0; s < SS; ++s) {
        const unsigned short* hin16 = (s & 1) ? h16b : h16a;
        const float*          hin32 = (s & 1) ? h32b : h32a;
        unsigned short*       hout16 = (s & 1) ? h16a : h16b;
        float*                hout32 = (s & 1) ? h32a : h32b;

        // pre_mm[b, n] = sum_k h[b,k] * U[n,k]
        v8f acc = {};
        const unsigned short* arow = hin16 + (long)b0 * HH;
        const unsigned short* brow = &Ul[nloc0 * LDSK];
#pragma unroll 4
        for (int k = 0; k < HH; k += 32) {
            v16bf a = load_frag_a(arow + k, HH, lane);
            v16bf b = load_frag_b(brow + k, LDSK, lane);
            acc = wmma_bf16(a, b, acc);
        }

        // Fused gate epilogue
#pragma unroll
        for (int r = 0; r < 8; ++r) {
            int  m    = b0 + mo + r;
            long oidx = ((long)m * SS + s) * HH + ncol;
            float pre = acc[r] + out[oidx];
            float z   = sigmoidf_(pre + bzv);
            float ht  = tanhf(pre + bhv);
            float hold = hin32[m * HH + ncol];
            float hn  = (zs * (1.0f - z) + ns) * ht + z * hold;
            out[oidx]              = hn;
            hout32[m * HH + ncol]  = hn;
            hout16[m * HH + ncol]  = f2bf(hn);
            if (s == SS - 1) hid_out[m * HH + ncol] = hn;
        }

        // Device-wide barrier between timesteps (monotone counter, no reset)
        __syncthreads();
        if (threadIdx.x == 0) {
            __hip_atomic_fetch_add(counter, 1u, __ATOMIC_RELEASE,
                                   __HIP_MEMORY_SCOPE_AGENT);
            unsigned target = (unsigned)NBLK * (unsigned)(s + 1);
            while (__hip_atomic_load(counter, __ATOMIC_ACQUIRE,
                                     __HIP_MEMORY_SCOPE_AGENT) < target) {
                __builtin_amdgcn_s_sleep(1);
            }
        }
        __syncthreads();
    }
}

// ---------------- launch ----------------
extern "C" void kernel_launch(void* const* d_in, const int* in_sizes, int n_in,
                              void* d_out, int out_size, void* d_ws, size_t ws_size,
                              hipStream_t stream) {
    const float* inp  = (const float*)d_in[0];   // [B,S,F]
    const float* W    = (const float*)d_in[1];   // [H,F]
    const float* U    = (const float*)d_in[2];   // [H,H]
    const float* bz   = (const float*)d_in[3];   // [H]
    const float* bh   = (const float*)d_in[4];   // [H]
    const float* zeta = (const float*)d_in[5];   // [1]
    const float* nu   = (const float*)d_in[6];   // [1]
    float* out = (float*)d_out;                  // [B*S*H] output + [B*H] hidden

    // Workspace layout (bytes)
    char* ws = (char*)d_ws;
    unsigned short* Xb   = (unsigned short*)(ws);              // 32 MB  (B*S*F bf16)
    unsigned short* Wb   = (unsigned short*)(ws + 33554432);   // 1 MB   (H*F bf16)
    unsigned short* Ub   = (unsigned short*)(ws + 34603008);   // 2 MB   (H*H bf16)
    float*          h32a = (float*)(ws + 36700160);            // 256 KB
    float*          h32b = (float*)(ws + 36962304);            // 256 KB
    unsigned short* h16a = (unsigned short*)(ws + 37224448);   // 128 KB
    unsigned short* h16b = (unsigned short*)(ws + 37355520);   // 128 KB
    unsigned*       cntr = (unsigned*)(ws + 37486592);         // 4 B

    // 1) bf16 conversions
    {
        int n4 = (BB * SS * FF) / 4;
        cvt_bf16_kernel<<<(n4 + 255) / 256, 256, 0, stream>>>(inp, Xb, n4);
        n4 = (HH * FF) / 4;
        cvt_bf16_kernel<<<(n4 + 255) / 256, 256, 0, stream>>>(W, Wb, n4);
        n4 = (HH * HH) / 4;
        cvt_bf16_kernel<<<(n4 + 255) / 256, 256, 0, stream>>>(U, Ub, n4);
    }

    // 2) h0 = 0, counter = 0
    init_kernel<<<(BB * HH + 255) / 256, 256, 0, stream>>>(h32a, h16a, cntr, BB * HH);

    // 3) Wx GEMM -> d_out[B,S,H]
    {
        dim3 grid((BB * SS) / 64, HH / 64);
        wx_gemm_kernel<<<grid, 256, 0, stream>>>(Xb, Wb, out);
    }

    // 4) persistent recurrent scan (U slice in LDS, grid barrier per step)
    {
        size_t lds = (size_t)64 * LDSK * sizeof(unsigned short);  // 132096 B
        fastgrnn_scan_kernel<<<NBLK, 128, lds, stream>>>(
            out, out + (size_t)BB * SS * HH, Ub, bz, bh, zeta, nu,
            h32a, h32b, h16a, h16b, cntr);
    }
}
